// ScaledDot_Production_7584912245262
// MI455X (gfx1250) — matrix operand order repaired and checked by reading, not compiled, not run on hardware
//
#include <hip/hip_runtime.h>

typedef __attribute__((ext_vector_type(16))) _Float16 v16h;
typedef __attribute__((ext_vector_type(8)))  _Float16 v8h;
typedef __attribute__((ext_vector_type(4)))  _Float16 v4h;
typedef __attribute__((ext_vector_type(8)))  float    v8f;

#define SEQ  2048
#define DH   64
#define BK   64        // keys processed per block iteration
#define NW   4         // waves per workgroup
#define QB   (NW * 16) // q rows per workgroup
#define LPAD 72        // padded LDS row length (halves): 144B rows, 16B-aligned subtiles
#define NCH  8         // float4 chunks staged per thread per tensor

union V16 { v16h v; v8h h[2]; };

// ---- 16-lane row reductions in pure VALU via DPP16 (no LDS traffic) ----
#define DPP_QX1 0xB1   // quad_perm [1,0,3,2]
#define DPP_QX2 0x4E   // quad_perm [2,3,0,1]
#define DPP_HM  0x141  // ROW_HALF_MIRROR
#define DPP_RM  0x140  // ROW_MIRROR
#define DPP_MAX(x, ctrl) \
  x = fmaxf(x, __int_as_float(__builtin_amdgcn_update_dpp(0, __float_as_int(x), ctrl, 0xF, 0xF, true)))
#define DPP_ADD(x, ctrl) \
  x = x + __int_as_float(__builtin_amdgcn_update_dpp(0, __float_as_int(x), ctrl, 0xF, 0xF, true))

__device__ __forceinline__ float row16_max(float x) {
  DPP_MAX(x, DPP_QX1); DPP_MAX(x, DPP_QX2); DPP_MAX(x, DPP_HM); DPP_MAX(x, DPP_RM);
  return x;
}
__device__ __forceinline__ float row16_sum(float x) {
  DPP_ADD(x, DPP_QX1); DPP_ADD(x, DPP_QX2); DPP_ADD(x, DPP_HM); DPP_ADD(x, DPP_RM);
  return x;
}

__global__ __launch_bounds__(NW * 32, 1)
void fa_fused_kernel(const float* __restrict__ Q, const float* __restrict__ K,
                     const float* __restrict__ V, float* __restrict__ O) {
  __shared__ _Float16 Kl[2][BK * LPAD];     // double-buffered K block [key][d]
  __shared__ _Float16 Vt[2][DH * LPAD];     // double-buffered V block, transposed [d][key]
  __shared__ _Float16 Pl[NW][16 * LPAD];    // per-wave P tile [qrow][key]

  const int tid  = threadIdx.x;
  const int lane = tid & 31;
  const int wv   = tid >> 5;
  const int col  = lane & 15;   // A/C row & B/C column index
  const int hf   = lane >> 4;   // lane half
  const int kbA  = hf * 8;      // A-operand K offset for this half (VGPR0-3)
  const int kbB  = hf * 8;      // B-operand K offset for this half

  const int  bh    = blockIdx.y;            // flattened (b,h)
  const long base  = (long)bh * SEQ * DH;
  const int  qrow0 = blockIdx.x * QB + wv * 16;

  // ---- resident Q tile as two 16x32 f16 A-operands ----
  // fold 1/sqrt(64) and log2(e) into the scale; softmax runs in exp2 space
  const float scale = 0.125f * 1.44269504088896340736f;
  v16h aQ0, aQ1;
  {
    const float* qp = Q + base + (long)(qrow0 + col) * DH;
#pragma unroll
    for (int e = 0; e < 8; ++e) {
      aQ0[e]     = (_Float16)(qp[kbA + e]           * scale);
      aQ0[e + 8] = (_Float16)(qp[kbA + 16 + e]      * scale);
      aQ1[e]     = (_Float16)(qp[32 + kbA + e]      * scale);
      aQ1[e + 8] = (_Float16)(qp[32 + kbA + 16 + e] * scale);
    }
  }

  const v8f vzero = {};
  v8f acc[4];
#pragma unroll
  for (int t = 0; t < 4; ++t) acc[t] = vzero;
  float mrun[8], lrun[8];
#pragma unroll
  for (int r = 0; r < 8; ++r) { mrun[r] = -1e30f; lrun[r] = 0.0f; }

  // ---- register staging of the next K/V block (16 x global_load_b128/thread) ----
  float4 kraw[NCH], vraw[NCH];
  {
    const float4* kp4 = (const float4*)(K + base);
    const float4* vp4 = (const float4*)(V + base);
#pragma unroll
    for (int j = 0; j < NCH; ++j) {
      kraw[j] = kp4[tid + j * 128];
      vraw[j] = vp4[tid + j * 128];
    }
  }

  int cur = 0;
  for (int kb = 0; kb < SEQ; kb += BK) {
    // ---- store staged regs (f32 -> f16) into LDS buffer `cur` ----
    _Float16* klc = Kl[cur];
    _Float16* vtc = Vt[cur];
#pragma unroll
    for (int j = 0; j < NCH; ++j) {
      const int c    = tid + j * 128;
      const int key  = c >> 4;
      const int dcol = (c & 15) * 4;
      v4h kh;
      kh[0] = (_Float16)kraw[j].x; kh[1] = (_Float16)kraw[j].y;
      kh[2] = (_Float16)kraw[j].z; kh[3] = (_Float16)kraw[j].w;
      *(v4h*)&klc[key * LPAD + dcol] = kh;
      vtc[(dcol + 0) * LPAD + key] = (_Float16)vraw[j].x;
      vtc[(dcol + 1) * LPAD + key] = (_Float16)vraw[j].y;
      vtc[(dcol + 2) * LPAD + key] = (_Float16)vraw[j].z;
      vtc[(dcol + 3) * LPAD + key] = (_Float16)vraw[j].w;
    }
    __syncthreads();   // single barrier per block: buf[cur] ready for all waves

    // ---- issue next block's global loads; consumed only next iteration ----
    if (kb + BK < SEQ) {
      const float4* kp4 = (const float4*)(K + base + (long)(kb + BK) * DH);
      const float4* vp4 = (const float4*)(V + base + (long)(kb + BK) * DH);
#pragma unroll
      for (int j = 0; j < NCH; ++j) {
        kraw[j] = kp4[tid + j * 128];
        vraw[j] = vp4[tid + j * 128];
      }
    }
    if (kb + 2 * BK < SEQ) {   // warm L2 two blocks ahead (global_prefetch_b8)
      __builtin_prefetch(&K[base + (long)(kb + 2 * BK) * DH + tid * 16], 0, 1);
      __builtin_prefetch(&V[base + (long)(kb + 2 * BK) * DH + tid * 16], 0, 1);
    }

    // ---- scores: 4 key sub-tiles of 16, contraction d=64 via 2 chained WMMAs ----
    v8f st[4];
#pragma unroll
    for (int nt = 0; nt < 4; ++nt) {
      const _Float16* kp = &klc[(nt * 16 + col) * LPAD];
      V16 b0, b1;
      b0.h[0] = *(const v8h*)(kp + kbB);
      b0.h[1] = *(const v8h*)(kp + kbB + 16);
      b1.h[0] = *(const v8h*)(kp + 32 + kbB);
      b1.h[1] = *(const v8h*)(kp + 32 + kbB + 16);
      v8f c = vzero;
      c = __builtin_amdgcn_wmma_f32_16x16x32_f16(false, aQ0, false, b0.v, (short)0, c, false, false);
      c = __builtin_amdgcn_wmma_f32_16x16x32_f16(false, aQ1, false, b1.v, (short)0, c, false, false);
      st[nt] = c;
    }

    // ---- online softmax over this 64-key block (row = r + 8*hf) ----
#pragma unroll
    for (int r = 0; r < 8; ++r) {
      float x = fmaxf(fmaxf(st[0][r], st[1][r]), fmaxf(st[2][r], st[3][r]));
      x = row16_max(x);
      const float mnew  = fmaxf(mrun[r], x);
      const float alpha = __builtin_amdgcn_exp2f(mrun[r] - mnew);
      mrun[r] = mnew;
      const int prow = (r + 8 * hf) * LPAD;
      float rs = 0.0f;
#pragma unroll
      for (int nt = 0; nt < 4; ++nt) {
        const float p = __builtin_amdgcn_exp2f(st[nt][r] - mnew);
        rs += p;
        Pl[wv][prow + nt * 16 + col] = (_Float16)p;
      }
      rs = row16_sum(rs);
      lrun[r] = lrun[r] * alpha + rs;
#pragma unroll
      for (int t = 0; t < 4; ++t) acc[t][r] *= alpha;
    }

    // ---- reload P in A-operand layout (same-wave LDS is in-order) ----
    V16 aP0, aP1;
    {
      const _Float16* pp = &Pl[wv][col * LPAD];
      aP0.h[0] = *(const v8h*)(pp + kbA);
      aP0.h[1] = *(const v8h*)(pp + kbA + 16);
      aP1.h[0] = *(const v8h*)(pp + 32 + kbA);
      aP1.h[1] = *(const v8h*)(pp + 32 + kbA + 16);
    }

    // ---- O += P @ V over 4 d-tiles; V^T rows give contiguous B-operand reads ----
#pragma unroll
    for (int dt = 0; dt < 4; ++dt) {
      const _Float16* vp = &vtc[(dt * 16 + col) * LPAD];
      V16 b0, b1;
      b0.h[0] = *(const v8h*)(vp + kbB);
      b0.h[1] = *(const v8h*)(vp + kbB + 16);
      b1.h[0] = *(const v8h*)(vp + 32 + kbB);
      b1.h[1] = *(const v8h*)(vp + 32 + kbB + 16);
      acc[dt] = __builtin_amdgcn_wmma_f32_16x16x32_f16(false, aP0.v, false, b0.v, (short)0, acc[dt], false, false);
      acc[dt] = __builtin_amdgcn_wmma_f32_16x16x32_f16(false, aP1.v, false, b1.v, (short)0, acc[dt], false, false);
    }

    cur ^= 1;
  }

  // ---- finalize: divide by row sums, store (16 contiguous floats per half-wave) ----
#pragma unroll
  for (int r = 0; r < 8; ++r) {
    const float inv = 1.0f / lrun[r];
    float* op = O + base + (long)(qrow0 + r + 8 * hf) * DH;
#pragma unroll
    for (int dt = 0; dt < 4; ++dt)
      op[dt * 16 + col] = acc[dt][r] * inv;
  }
}

extern "C" void kernel_launch(void* const* d_in, const int* in_sizes, int n_in,
                              void* d_out, int out_size, void* d_ws, size_t ws_size,
                              hipStream_t stream) {
  (void)in_sizes; (void)n_in; (void)out_size; (void)d_ws; (void)ws_size;
  const float* q = (const float*)d_in[0];
  const float* k = (const float*)d_in[1];
  const float* v = (const float*)d_in[2];
  float*       o = (float*)d_out;
  dim3 grid(SEQ / QB, 4 * 16);   // (q-blocks, B*H)
  fa_fused_kernel<<<grid, NW * 32, 0, stream>>>(q, k, v, o);
}
